// TransformerModel_31928786878805
// MI455X (gfx1250) — matrix operand order repaired and checked by reading, not compiled, not run on hardware
//
#include <hip/hip_runtime.h>

// ---------------------------------------------------------------------------
// MI455X (gfx1250, wave32) implementation of the top-k cross-attention model.
// GEMMs use v_wmma_f32_16x16x32_bf16 (16x16 C tiles, K=32/instr).
// All WMMA B operands are pre-transposed so every fragment load is a single
// contiguous 32-byte vector load per lane (no u16 gathers).
// Requires ws_size >= 230 MB (buffers are lifetime-overlapped).
// ---------------------------------------------------------------------------

typedef __bf16 bf16_t;
typedef __attribute__((ext_vector_type(16))) __bf16 v16bf;
typedef __attribute__((ext_vector_type(8)))  __bf16 v8bf;
typedef __attribute__((ext_vector_type(8)))  float  v8f;

#define BB 16
#define LL 2048
#define SS 128
#define HH 512
#define NH 8
#define EE 64

__device__ __forceinline__ bf16_t f2bf(float f) {
    union { float f; unsigned u; } uf; uf.f = f;
    unsigned r = uf.u + 0x7FFFu + ((uf.u >> 16) & 1u);   // round-to-nearest-even
    union { unsigned short s; bf16_t b; } ub; ub.s = (unsigned short)(r >> 16);
    return ub.b;
}

__device__ __forceinline__ v8f wmma_bf16(v16bf a, v16bf b, v8f c) {
    // (neg_a, A, neg_b, B, c_mod, C, reuse_a, reuse_b)
    return __builtin_amdgcn_wmma_f32_16x16x32_bf16(false, a, false, b, (short)0, c,
                                                   false, false);
}

// A fragment, 16x32 (MxK) bf16, row-major source, base at tile origin.
// lane<16: rows 0-15, K 0..15 ; lane>=16: rows 0-15, K 16..31
__device__ __forceinline__ v16bf load_a_frag(const bf16_t* __restrict__ base,
                                             int lda, int lane) {
    int m    = lane & 15;
    int koff = (lane >> 4) << 4;            // 0 or 16
    const bf16_t* rp = base + (size_t)m * lda + koff;
    union { v16bf v; v8bf h[2]; } af;
    af.h[0] = *(const v8bf*)(rp);           // K = koff .. koff+7
    af.h[1] = *(const v8bf*)(rp + 8);       // K = koff+8 .. koff+15
    return af.v;
}

// B fragment, 32x16 (KxN), where the source holds B^T row-major [N,K]
// (transposed weights, K-matrix, transposed V). One 32B load per lane.
// lane = column n (lane&15); K 0-15 in lanes 0-15, K 16-31 in lanes 16-31.
__device__ __forceinline__ v16bf load_b_tn(const bf16_t* __restrict__ base,
                                           int lds, int lane) {
    int n  = lane & 15;
    int kb = (lane >> 4) << 4;
    return *(const v16bf*)(base + (size_t)n * lds + kb);
}

// ---------------------------------------------------------------------------
// Transpose + convert f32 W[K,N] -> bf16 WT[N,K]  (512x512)
__global__ void k_cvt_t(const float* __restrict__ in, bf16_t* __restrict__ out) {
    int i = blockIdx.x * blockDim.x + threadIdx.x;   // coalesced read
    int k = i >> 9, n = i & 511;
    out[(size_t)n * HH + k] = f2bf(in[i]);
}

// L2-normalize over the sequence dimension (axis=1): one block per batch,
// thread d owns feature column d (coalesced across threads per step).
__global__ void k_colnorm(const float* __restrict__ in, bf16_t* __restrict__ out,
                          int Lseq) {
    int b = blockIdx.x, d = threadIdx.x;                  // blockDim = 512
    const float* p = in + (size_t)b * Lseq * HH + d;
    float acc = 0.f;
    for (int l = 0; l < Lseq; ++l) { float v = p[(size_t)l * HH]; acc += v * v; }
    float inv = 1.0f / fmaxf(sqrtf(acc), 1e-12f);
    bf16_t* q = out + (size_t)b * Lseq * HH + d;
    for (int l = 0; l < Lseq; ++l) q[(size_t)l * HH] = f2bf(p[(size_t)l * HH] * inv);
}

// C[M,512] = A[M,512] @ W + bias (W given transposed: WT[512,512] = [N,K]).
// bf16 in / bf16 out, f32 accum.
// grid = (512/64, M/128); block = 256 (8 waves); wave -> one 16x64 slab.
__global__ void k_proj(const bf16_t* __restrict__ A, const bf16_t* __restrict__ WT,
                       const float* __restrict__ bias, bf16_t* __restrict__ C) {
    int lane = threadIdx.x & 31, wave = threadIdx.x >> 5;
    int row0 = (blockIdx.y * 8 + wave) * 16;
    int col0 = blockIdx.x * 64;
    v8f acc[4] = {};
    for (int k0 = 0; k0 < HH; k0 += 32) {
        const bf16_t* ap = A + (size_t)row0 * HH + k0;
        __builtin_prefetch(ap + 32, 0, 1);                 // global_prefetch_b8
        v16bf a = load_a_frag(ap, HH, lane);
#pragma unroll
        for (int t = 0; t < 4; ++t) {
            v16bf bm = load_b_tn(WT + (size_t)(col0 + 16 * t) * HH + k0, HH, lane);
            acc[t] = wmma_bf16(a, bm, acc[t]);
        }
    }
    int cl = lane & 15, rh = (lane >> 4) * 8;
#pragma unroll
    for (int t = 0; t < 4; ++t) {
        int c = col0 + 16 * t + cl;
        float bv = bias[c];
#pragma unroll
        for (int r = 0; r < 8; ++r)
            C[(size_t)(row0 + rh + r) * HH + c] = f2bf(acc[t][r] + bv);
    }
}

// Same GEMM, but stores the result head-transposed for the AV stage:
// VT[(b*NH + h) * EE * Sv + e * Sv + s],  GEMM row = b*Sv + s, col = h*64 + e.
__global__ void k_proj_t(const bf16_t* __restrict__ A, const bf16_t* __restrict__ WT,
                         const float* __restrict__ bias, bf16_t* __restrict__ VT,
                         int Sv) {
    int lane = threadIdx.x & 31, wave = threadIdx.x >> 5;
    int row0 = (blockIdx.y * 8 + wave) * 16;
    int col0 = blockIdx.x * 64;
    v8f acc[4] = {};
    for (int k0 = 0; k0 < HH; k0 += 32) {
        const bf16_t* ap = A + (size_t)row0 * HH + k0;
        __builtin_prefetch(ap + 32, 0, 1);
        v16bf a = load_a_frag(ap, HH, lane);
#pragma unroll
        for (int t = 0; t < 4; ++t) {
            v16bf bm = load_b_tn(WT + (size_t)(col0 + 16 * t) * HH + k0, HH, lane);
            acc[t] = wmma_bf16(a, bm, acc[t]);
        }
    }
    int cl = lane & 15, rh = (lane >> 4) * 8;
    int b  = row0 / Sv;                       // 16 | Sv, so one batch per tile
    int s0 = row0 - b * Sv;
    int h  = col0 >> 6;
    bf16_t* Vb = VT + ((size_t)(b * NH + h) * EE) * Sv;
#pragma unroll
    for (int t = 0; t < 4; ++t) {
        int e = 16 * t + cl;                  // col0 is 64-aligned
        float bv = bias[col0 + 16 * t + cl];
#pragma unroll
        for (int r = 0; r < 8; ++r)
            Vb[(size_t)e * Sv + s0 + rh + r] = f2bf(acc[t][r] + bv);
    }
}

// scores[b,h,lq,s] = Q[b,lq,h,:] . K[b,s,h,:]  (E=64, 2 WMMA K-steps)
// grid = (S/64, Lq/128, B*NH)
__global__ void k_qk(const bf16_t* __restrict__ Q, const bf16_t* __restrict__ Kt,
                     float* __restrict__ sc, int Lq, int S) {
    int lane = threadIdx.x & 31, wave = threadIdx.x >> 5;
    int bh = blockIdx.z, b = bh >> 3, h = bh & 7;
    int row0 = (blockIdx.y * 8 + wave) * 16;
    int col0 = blockIdx.x * 64;
    const bf16_t* Qb = Q + (size_t)b * Lq * HH + (size_t)h * EE;
    const bf16_t* Kb = Kt + (size_t)b * S * HH + (size_t)h * EE;
    float* Sb = sc + (size_t)bh * Lq * S;
    v8f acc[4] = {};
#pragma unroll
    for (int k0 = 0; k0 < EE; k0 += 32) {
        v16bf a = load_a_frag(Qb + (size_t)row0 * HH + k0, HH, lane);
#pragma unroll
        for (int t = 0; t < 4; ++t) {
            v16bf bm = load_b_tn(Kb + (size_t)(col0 + 16 * t) * HH + k0, HH, lane);
            acc[t] = wmma_bf16(a, bm, acc[t]);
        }
    }
    int cl = lane & 15, rh = (lane >> 4) * 8;
#pragma unroll
    for (int t = 0; t < 4; ++t)
#pragma unroll
        for (int r = 0; r < 8; ++r)
            Sb[(size_t)(row0 + rh + r) * S + col0 + 16 * t + cl] = acc[t][r];
}

// out[b,lq,h,:] = Amat[b,h,lq,:] @ V[b,:,h,:] with V pre-transposed:
// VT[(b*NH+h)*EE*S + e*S + s].   grid = (1, Lq/128, B*NH), N=64 = one wave slab.
__global__ void k_av(const bf16_t* __restrict__ Amat, const bf16_t* __restrict__ VT,
                     float* __restrict__ out, int Lq, int S) {
    int lane = threadIdx.x & 31, wave = threadIdx.x >> 5;
    int bh = blockIdx.z, b = bh >> 3, h = bh & 7;
    int row0 = (blockIdx.y * 8 + wave) * 16;
    const bf16_t* Ab  = Amat + (size_t)bh * Lq * S;
    const bf16_t* VTb = VT + (size_t)bh * EE * S;
    float* Ob = out + (size_t)b * Lq * HH + (size_t)h * EE;
    v8f acc[4] = {};
    for (int k0 = 0; k0 < S; k0 += 32) {
        const bf16_t* ap = Ab + (size_t)row0 * S + k0;
        __builtin_prefetch(ap + 32, 0, 1);
        v16bf a = load_a_frag(ap, S, lane);
#pragma unroll
        for (int t = 0; t < 4; ++t) {
            v16bf bm = load_b_tn(VTb + (size_t)(16 * t) * S + k0, S, lane);
            acc[t] = wmma_bf16(a, bm, acc[t]);
        }
    }
    int cl = lane & 15, rh = (lane >> 4) * 8;
#pragma unroll
    for (int t = 0; t < 4; ++t)
#pragma unroll
        for (int r = 0; r < 8; ++r)
            Ob[(size_t)(row0 + rh + r) * HH + 16 * t + cl] = acc[t][r];
}

// One wave32 per row: select top-32 of S=CPL*32, masked softmax (temp folded).
template <int CPL>
__global__ void k_topk_softmax(const float* __restrict__ sc,
                               bf16_t* __restrict__ Aout, float temp) {
    const int S = CPL * 32;
    int lane = threadIdx.x & 31;
    int row = blockIdx.x * (blockDim.x >> 5) + (threadIdx.x >> 5);
    const float* base = sc + (size_t)row * S;
    float v[CPL];
#pragma unroll
    for (int i = 0; i < CPL; ++i) v[i] = base[i * 32 + lane];
    unsigned long long selmask = 0ull;
    float m0 = 0.f;
#pragma unroll 1
    for (int it = 0; it < 32; ++it) {
        float m = -3.4e38f;
#pragma unroll
        for (int i = 0; i < CPL; ++i)
            if (!((selmask >> i) & 1ull)) m = fmaxf(m, v[i]);
#pragma unroll
        for (int off = 16; off >= 1; off >>= 1)
            m = fmaxf(m, __shfl_xor(m, off, 32));
        if (it == 0) m0 = m;
        int mi = -1;
#pragma unroll
        for (int i = CPL - 1; i >= 0; --i)
            if (!((selmask >> i) & 1ull) && v[i] == m) mi = i;   // smallest i wins
        unsigned long long bal = __ballot(mi >= 0);
        if (bal != 0ull) {
            int leader = (int)__builtin_ctzll(bal);
            if (lane == leader) selmask |= (1ull << mi);
        }
    }
    float s = 0.f;
#pragma unroll
    for (int i = 0; i < CPL; ++i) {
        float p = ((selmask >> i) & 1ull) ? __expf(temp * (v[i] - m0)) : 0.f;
        v[i] = p; s += p;
    }
#pragma unroll
    for (int off = 16; off >= 1; off >>= 1) s += __shfl_xor(s, off, 32);
    float inv = 1.f / s;
    bf16_t* ob = Aout + (size_t)row * S;
#pragma unroll
    for (int i = 0; i < CPL; ++i) ob[i * 32 + lane] = f2bf(v[i] * inv);
}

// logits[row] = X[row,:] . w_pool + b_pool ; one wave per row.
__global__ void k_pool_logits(const float* __restrict__ X,
                              const float* __restrict__ wp,
                              const float* __restrict__ bp,
                              float* __restrict__ logits) {
    int lane = threadIdx.x & 31;
    int row = blockIdx.x * (blockDim.x >> 5) + (threadIdx.x >> 5);
    const float* xb = X + (size_t)row * HH;
    float acc = 0.f;
#pragma unroll
    for (int i = 0; i < 16; ++i) { int c = i * 32 + lane; acc += xb[c] * wp[c]; }
#pragma unroll
    for (int off = 16; off >= 1; off >>= 1) acc += __shfl_xor(acc, off, 32);
    if (lane == 0) logits[row] = acc + bp[0];
}

// softmax over n per block(=batch)
__global__ void k_row_softmax(const float* __restrict__ logits,
                              float* __restrict__ attn, int n) {
    __shared__ float red[256];
    int b = blockIdx.x, t = threadIdx.x;
    const float* lb = logits + (size_t)b * n;
    float m = -3.4e38f;
    for (int i = t; i < n; i += 256) m = fmaxf(m, lb[i]);
    red[t] = m; __syncthreads();
    for (int s = 128; s; s >>= 1) { if (t < s) red[t] = fmaxf(red[t], red[t + s]); __syncthreads(); }
    m = red[0]; __syncthreads();
    float sum = 0.f;
    for (int i = t; i < n; i += 256) sum += __expf(lb[i] - m);
    red[t] = sum; __syncthreads();
    for (int s = 128; s; s >>= 1) { if (t < s) red[t] += red[t + s]; __syncthreads(); }
    float inv = 1.f / red[0];
    for (int i = t; i < n; i += 256) attn[(size_t)b * n + i] = __expf(lb[i] - m) * inv;
}

// pooled[b,d] = sum_l attn[b,l] * X[b,l,d] ; block per b, 512 threads.
__global__ void k_pool_weighted(const float* __restrict__ X,
                                const float* __restrict__ attn,
                                float* __restrict__ pooled, int n) {
    int b = blockIdx.x, d = threadIdx.x;
    const float* xb = X + (size_t)b * n * HH + d;
    const float* ab = attn + (size_t)b * n;
    float acc = 0.f;
    for (int l = 0; l < n; ++l) acc += ab[l] * xb[(size_t)l * HH];
    pooled[b * HH + d] = acc;
}

// emb = l2norm(concat(p1,p2)) over features; written to d_out and ws copy.
__global__ void k_embnorm(const float* __restrict__ p1, const float* __restrict__ p2,
                          float* __restrict__ emb_out, float* __restrict__ embN) {
    __shared__ float red[256];
    int b = blockIdx.x, t = threadIdx.x;
    float vals[4], loc = 0.f;
#pragma unroll
    for (int j = 0; j < 4; ++j) {
        int f = t + j * 256;
        float v = (f < HH) ? p1[b * HH + f] : p2[b * HH + f - HH];
        vals[j] = v; loc += v * v;
    }
    red[t] = loc; __syncthreads();
    for (int s = 128; s; s >>= 1) { if (t < s) red[t] += red[t + s]; __syncthreads(); }
    float inv = 1.f / fmaxf(sqrtf(red[0]), 1e-12f);
#pragma unroll
    for (int j = 0; j < 4; ++j) {
        int f = t + j * 256;
        float o = vals[j] * inv;
        emb_out[b * 1024 + f] = o;
        embN[b * 1024 + f] = o;
    }
}

// tiny fp32 FC: Y[m,n] = act(X[m,:] @ W[:,n] + b[n])
__global__ void k_fc(const float* __restrict__ X, const float* __restrict__ W,
                     const float* __restrict__ bias, float* __restrict__ Y,
                     int M, int K, int N, int relu) {
    int id = blockIdx.x * blockDim.x + threadIdx.x;
    if (id >= M * N) return;
    int m = id / N, n = id % N;
    float acc = bias[n];
    for (int k = 0; k < K; ++k) acc += X[(size_t)m * K + k] * W[(size_t)k * N + n];
    Y[id] = relu ? fmaxf(acc, 0.f) : acc;
}

// ---------------------------------------------------------------------------
extern "C" void kernel_launch(void* const* d_in, const int* in_sizes, int n_in,
                              void* d_out, int out_size, void* d_ws, size_t ws_size,
                              hipStream_t stream) {
    (void)in_sizes; (void)n_in; (void)out_size; (void)ws_size;
    const float* seqs  = (const float*)d_in[0];
    const float* smis  = (const float*)d_in[1];
    const float* Wq    = (const float*)d_in[2];
    const float* bq    = (const float*)d_in[3];
    const float* Wk    = (const float*)d_in[4];
    const float* bk    = (const float*)d_in[5];
    const float* Wv    = (const float*)d_in[6];
    const float* bv    = (const float*)d_in[7];
    const float* wpool = (const float*)d_in[8];
    const float* bpool = (const float*)d_in[9];
    const float* W1    = (const float*)d_in[10];
    const float* b1    = (const float*)d_in[11];
    const float* W2    = (const float*)d_in[12];
    const float* b2    = (const float*)d_in[13];
    const float* W3    = (const float*)d_in[14];
    const float* b3    = (const float*)d_in[15];

    float* outp  = (float*)d_out;
    float* emb_o = outp;                 // [16,1024]
    float* out_o = outp + 16384;         // [16]
    float* attn1 = outp + 16400;         // [16,2048]
    float* attn2 = outp + 49168;         // [16,128]

    const size_t MB = 1024u * 1024u;
    char* ws = (char*)d_ws;
    // persistent
    bf16_t* seqsN = (bf16_t*)(ws + 0);              // 32 MB
    bf16_t* smisN = (bf16_t*)(ws + 32 * MB);        // 2 MB
    bf16_t* WqT   = (bf16_t*)(ws + 34 * MB);        // 512 KB each, transposed
    bf16_t* WkT   = (bf16_t*)(ws + 34 * MB + 512 * 1024);
    bf16_t* WvT   = (bf16_t*)(ws + 35 * MB);
    float*  logit = (float*)(ws + 35 * MB + 512 * 1024);  // <=128 KB
    float*  pool1 = logit + BB * LL;                       // [16,512]
    float*  pool2 = pool1 + BB * HH;
    float*  embN  = pool2 + BB * HH;                       // [16,1024]
    float*  h1    = embN + BB * 1024;
    float*  h2    = h1 + BB * 1024;
    // arena (lifetime-overlapped; sequential kernels on one stream)
    bf16_t* Qs   = (bf16_t*)(ws + 36 * MB);    // 32 MB  (dead after qk1)
    float*  out1 = (float*)(ws + 36 * MB);     // 64 MB  (written in av1)
    float*  sc1  = (float*)(ws + 68 * MB);     // 128 MB  (dead after topk1)
    bf16_t* A1   = (bf16_t*)(ws + 100 * MB);   // 64 MB
    bf16_t* Kx   = (bf16_t*)(ws + 108 * MB);   // 2 MB
    bf16_t* VxT  = (bf16_t*)(ws + 110 * MB);   // 2 MB  (head-transposed V)
    bf16_t* Qx   = (bf16_t*)(ws + 36 * MB);    // 2 MB   (after pool1)
    bf16_t* Ks   = (bf16_t*)(ws + 38 * MB);    // 32 MB
    bf16_t* VsT  = (bf16_t*)(ws + 70 * MB);    // 32 MB (head-transposed V)
    float*  sc2  = (float*)(ws + 102 * MB);    // 128 MB
    bf16_t* A2   = (bf16_t*)(ws + 118 * MB);   // 64 MB
    float*  out2 = (float*)(ws + 126 * MB);    // 4 MB  -> peak ~230 MB

    const float temp = 0.125f;   // 1/sqrt(64)

    // 0) weights -> transposed bf16 (B operands of all projections)
    k_cvt_t<<<(HH * HH) / 256, 256, 0, stream>>>(Wq, WqT);
    k_cvt_t<<<(HH * HH) / 256, 256, 0, stream>>>(Wk, WkT);
    k_cvt_t<<<(HH * HH) / 256, 256, 0, stream>>>(Wv, WvT);
    // 1) L2 norm over sequence dim
    k_colnorm<<<BB, HH, 0, stream>>>(seqs, seqsN, LL);
    k_colnorm<<<BB, HH, 0, stream>>>(smis, smisN, SS);
    // 2) fusion 1 projections  (q from protein, k/v from drug)
    k_proj<<<dim3(8, (BB * LL) / 128), 256, 0, stream>>>(seqsN, WqT, bq, Qs);
    k_proj<<<dim3(8, (BB * SS) / 128), 256, 0, stream>>>(smisN, WkT, bk, Kx);
    k_proj_t<<<dim3(8, (BB * SS) / 128), 256, 0, stream>>>(smisN, WvT, bv, VxT, SS);
    // 3) QK1 [B,H,2048,128]
    k_qk<<<dim3(SS / 64, LL / 128, BB * NH), 256, 0, stream>>>(Qs, Kx, sc1, LL, SS);
    // 4) top-32 + softmax -> A1 (bf16)
    k_topk_softmax<SS / 32><<<(BB * NH * LL) / 8, 256, 0, stream>>>(sc1, A1, temp);
    // 5) AV1 -> smiles_out [B,2048,512]
    k_av<<<dim3(1, LL / 128, BB * NH), 256, 0, stream>>>(A1, VxT, out1, LL, SS);
    // 6) attention pool over L
    k_pool_logits<<<(BB * LL) / 8, 256, 0, stream>>>(out1, wpool, bpool, logit);
    k_row_softmax<<<BB, 256, 0, stream>>>(logit, attn1, LL);
    k_pool_weighted<<<BB, HH, 0, stream>>>(out1, attn1, pool1, LL);
    // 7) fusion 2 projections  (q from drug, k/v from protein)
    k_proj<<<dim3(8, (BB * SS) / 128), 256, 0, stream>>>(smisN, WqT, bq, Qx);
    k_proj<<<dim3(8, (BB * LL) / 128), 256, 0, stream>>>(seqsN, WkT, bk, Ks);
    k_proj_t<<<dim3(8, (BB * LL) / 128), 256, 0, stream>>>(seqsN, WvT, bv, VsT, LL);
    // 8) QK2 [B,H,128,2048]
    k_qk<<<dim3(LL / 64, SS / 128, BB * NH), 256, 0, stream>>>(Qx, Ks, sc2, SS, LL);
    // 9) top-32 + softmax over 2048 -> A2
    k_topk_softmax<LL / 32><<<(BB * NH * SS) / 8, 256, 0, stream>>>(sc2, A2, temp);
    // 10) AV2 -> seqs_out [B,128,512]
    k_av<<<dim3(1, SS / 128, BB * NH), 256, 0, stream>>>(A2, VsT, out2, SS, LL);
    // 11) attention pool over S
    k_pool_logits<<<(BB * SS) / 8, 256, 0, stream>>>(out2, wpool, bpool, logit);
    k_row_softmax<<<BB, 256, 0, stream>>>(logit, attn2, SS);
    k_pool_weighted<<<BB, HH, 0, stream>>>(out2, attn2, pool2, SS);
    // 12) emb = l2norm(concat) ; 13) MLP head
    k_embnorm<<<BB, 256, 0, stream>>>(pool1, pool2, emb_o, embN);
    k_fc<<<(BB * 1024 + 255) / 256, 256, 0, stream>>>(embN, W1, b1, h1, BB, 1024, 1024, 1);
    k_fc<<<(BB * 512 + 255) / 256, 256, 0, stream>>>(h1, W2, b2, h2, BB, 1024, 512, 1);
    k_fc<<<1, 64, 0, stream>>>(h2, W3, b3, out_o, BB, 512, 1, 0);
}
